// MIPSBruteForceTopK_90907277787716
// MI455X (gfx1250) — matrix operand order repaired and checked by reading, not compiled, not run on hardware
//
#include <hip/hip_runtime.h>
#include <hip/hip_bf16.h>

// B=128 queries, D=128, N=1e6 items, K=100.
// Pass 1: TDM-staged bf16x3-split WMMA GEMM -> scores (128 x 1e6 fp32) in d_ws.
// Pass 2: per-query exact streaming top-100 with deterministic tie-breaking.

typedef __attribute__((ext_vector_type(16))) __bf16        v16bf;
typedef __attribute__((ext_vector_type(8)))  float         v8f;
typedef __attribute__((ext_vector_type(8)))  unsigned int  v8u;
typedef __attribute__((ext_vector_type(4)))  unsigned int  u32x4;
typedef __attribute__((ext_vector_type(8)))  unsigned int  u32x8;

typedef __attribute__((address_space(3))) float LDSfloat;

#define NQ     128
#define ND     128
#define NITEMS 1000000
#define TOPK   100
#define AROW   136               // padded bf16 row stride for LDS A
#define BROW   66                // B tile LDS row stride in f32 (64 data + 2 pad)
#define SBF_BYTES (ND * BROW * 4)  // 33792 bytes for f32 B tile
#define CHUNK  2048
#define BT2    512

__device__ __forceinline__ unsigned int f32_to_bf16_rn(float x) {
  unsigned int u = __float_as_uint(x);
  return (u + 0x7FFFu + ((u >> 16) & 1u)) >> 16;   // round-to-nearest-even
}

// ---------------------------------------------------------------------------
// Kernel 1: scores[q][n] = sum_k Q[q][k] * Tt[k][n]  via bf16x3 WMMA.
// grid = NITEMS/64 blocks of 128 threads (4 waves); each wave: 16 items x 128 q.
// B tile (128 K x 64 items, fp32, row stride 1e6) staged to LDS by one
// tensor_load_to_lds DMA with pad_interval=64DW/pad_amount=2DW -> stride 66 DW.
// ---------------------------------------------------------------------------
__global__ void __launch_bounds__(128)
mips_gemm_scores_kernel(const float* __restrict__ Q,
                        const float* __restrict__ Tt,
                        float* __restrict__ S) {
  extern __shared__ unsigned char smraw[];
  float* sBf = (float*)smraw;                                  // [128][BROW] f32
  unsigned short* sAhi = (unsigned short*)(smraw + SBF_BYTES); // [128][AROW] bf16 hi
  unsigned short* sAlo = sAhi + NQ * AROW;                     // [128][AROW] bf16 lo

  const int tid = threadIdx.x;

  // ---- TDM: DMA the 128x64 f32 item tile into LDS (wave 0 issues).
  if (tid < 32) {
    unsigned long long gaddr =
        (unsigned long long)(const void*)(Tt + (size_t)blockIdx.x * 64);
    unsigned lds_off = (unsigned)(unsigned long long)(LDSfloat*)sBf;

    u32x4 g0;
    g0[0] = 1u;                                   // count=1, user descriptor
    g0[1] = lds_off;                              // lds_addr
    g0[2] = (unsigned)(gaddr & 0xFFFFFFFFull);    // global_addr[31:0]
    g0[3] = (unsigned)((gaddr >> 32) & 0x1FFFFFFu) | 0x80000000u; // [56:32]|type=2

    u32x8 g1;
    g1[0] = (2u << 16)      // data_size = 4 bytes
          | (1u << 20)      // pad_enable
          | (5u << 22)      // pad_interval: 64 DWORDs
          | (1u << 25);     // pad_amount: 2 DWORDs
    g1[1] = ((unsigned)NITEMS & 0xFFFFu) << 16;           // tensor_dim0[15:0]
    g1[2] = ((unsigned)NITEMS >> 16) | (128u << 16);      // dim0[31:16] | dim1[15:0]
    g1[3] = (64u << 16);                                  // dim1[31:16]=0 | tile_dim0=64
    g1[4] = 128u;                                         // tile_dim1=128, tile_dim2=0
    g1[5] = (unsigned)NITEMS;                             // tensor_dim0_stride[31:0]
    g1[6] = 0u;                                           // stride[47:32] | dim1_stride lo
    g1[7] = 0u;
    asm volatile("tensor_load_to_lds %0, %1" :: "s"(g0), "s"(g1) : "memory");
  }

  // ---- Stage A = Q (128x128 f32) into LDS as bf16 hi/lo while TDM runs.
  for (int i = tid; i < NQ * ND; i += 128) {
    int m = i >> 7, k = i & 127;
    float x = Q[i];
    unsigned int hb = f32_to_bf16_rn(x);
    float hf = __uint_as_float(hb << 16);
    unsigned int lb = f32_to_bf16_rn(x - hf);
    sAhi[m * AROW + k] = (unsigned short)hb;
    sAlo[m * AROW + k] = (unsigned short)lb;
  }
  if (tid < 32) __builtin_amdgcn_s_wait_tensorcnt(0);
  __syncthreads();

  const int lane  = tid & 31;
  const int wave  = tid >> 5;
  const int col   = lane & 15;     // N (and M) position within tile
  const int lhalf = lane >> 4;     // lane half selects contiguous K-range 16*lhalf..+15
  const int ci    = wave * 16 + col;                       // item col in LDS tile
  const int c     = blockIdx.x * 64 + ci;                  // global item column

  v8f acc[8];
#pragma unroll
  for (int m = 0; m < 8; ++m)
#pragma unroll
    for (int v = 0; v < 8; ++v) acc[m][v] = 0.0f;

  for (int k0 = 0; k0 < ND; k0 += 32) {
    // ---- B fragment (32x16 bf16): lane<16 -> K = k0..k0+15, lane>=16 -> +16
    const int kb = k0 + lhalf * 16;
    v8u bhiU, bloU;
#pragma unroll
    for (int r = 0; r < 8; ++r) {
      float e0 = sBf[(kb + 2 * r) * BROW + ci];
      float e1 = sBf[(kb + 2 * r + 1) * BROW + ci];
      unsigned int h0 = f32_to_bf16_rn(e0);
      unsigned int h1 = f32_to_bf16_rn(e1);
      float f0 = __uint_as_float(h0 << 16);
      float f1 = __uint_as_float(h1 << 16);
      unsigned int l0 = f32_to_bf16_rn(e0 - f0);
      unsigned int l1 = f32_to_bf16_rn(e1 - f1);
      bhiU[r] = h0 | (h1 << 16);
      bloU[r] = l0 | (l1 << 16);
    }
    v16bf bhi = __builtin_bit_cast(v16bf, bhiU);
    v16bf blo = __builtin_bit_cast(v16bf, bloU);

    const int kA = k0 + lhalf * 16;  // A fragment K base (contiguous, matches B)
#pragma unroll
    for (int m = 0; m < 8; ++m) {
      const int M = m * 16 + col;
      const unsigned int* rowHi = (const unsigned int*)(sAhi + M * AROW);
      const unsigned int* rowLo = (const unsigned int*)(sAlo + M * AROW);
      v8u ahiU, aloU;
#pragma unroll
      for (int r = 0; r < 4; ++r) {
        ahiU[r]     = rowHi[(kA >> 1) + r];        // K = kA+2r, kA+2r+1
        ahiU[r + 4] = rowHi[(kA >> 1) + 4 + r];    // K = kA+8+2r, ...
        aloU[r]     = rowLo[(kA >> 1) + r];
        aloU[r + 4] = rowLo[(kA >> 1) + 4 + r];
      }
      v16bf ahi = __builtin_bit_cast(v16bf, ahiU);
      v16bf alo = __builtin_bit_cast(v16bf, aloU);

      acc[m] = __builtin_amdgcn_wmma_f32_16x16x32_bf16(false, ahi, false, bhi,
                                                       (short)0, acc[m], false, false);
      acc[m] = __builtin_amdgcn_wmma_f32_16x16x32_bf16(false, ahi, false, blo,
                                                       (short)0, acc[m], false, false);
      acc[m] = __builtin_amdgcn_wmma_f32_16x16x32_bf16(false, alo, false, bhi,
                                                       (short)0, acc[m], false, false);
    }
  }

  // Store: 32-bit C/D layout: lanes 0-15 -> M = v, lanes 16-31 -> M = v+8
#pragma unroll
  for (int m = 0; m < 8; ++m)
#pragma unroll
    for (int v = 0; v < 8; ++v) {
      int row = m * 16 + lhalf * 8 + v;
      S[(size_t)row * NITEMS + c] = acc[m][v];
    }
}

// ---------------------------------------------------------------------------
// Kernel 2: per-query exact top-100 (sorted desc, ties -> lower index)
// ---------------------------------------------------------------------------
__device__ __forceinline__ bool better(float va, int ia, float vb, int ib) {
  return (va > vb) || (va == vb && ia < ib);
}

__global__ void __launch_bounds__(BT2)
mips_topk_kernel(const float* __restrict__ S,
                 const int* __restrict__ ids,
                 float* __restrict__ out) {
  __shared__ float qv[CHUNK];
  __shared__ int   qi[CHUNK];
  __shared__ float topv[128];
  __shared__ int   topi[128];
  __shared__ int   cnt;

  const int q   = blockIdx.x;
  const int tid = threadIdx.x;
  const float* rowp = S + (size_t)q * NITEMS;

  // ---- seed: bitonic-sort first CHUNK scores (desc, tie -> lower idx)
  for (int i = tid; i < CHUNK; i += BT2) { qv[i] = rowp[i]; qi[i] = i; }
  __syncthreads();
  for (int kk = 2; kk <= CHUNK; kk <<= 1) {
    for (int j = kk >> 1; j > 0; j >>= 1) {
      for (int i = tid; i < CHUNK; i += BT2) {
        int ixj = i ^ j;
        if (ixj > i) {
          float va = qv[i], vb = qv[ixj];
          int   ia = qi[i], ib = qi[ixj];
          bool aB = better(va, ia, vb, ib);
          bool descHere = ((i & kk) == 0);
          if (descHere ? !aB : aB) {
            qv[i] = vb; qv[ixj] = va;
            qi[i] = ib; qi[ixj] = ia;
          }
        }
      }
      __syncthreads();
    }
  }
  if (tid < 128) { topv[tid] = qv[tid]; topi[tid] = qi[tid]; }
  if (tid == 0) cnt = 0;
  __syncthreads();

  // ---- stream remaining items
  for (int base = CHUNK; base < NITEMS; base += CHUNK) {
    float thrV = topv[TOPK - 1];
    int   thrI = topi[TOPK - 1];
    __builtin_prefetch(rowp + base + CHUNK, 0, 0);
    for (int u = 0; u < CHUNK; u += BT2) {
      int idx = base + u + tid;
      if (idx < NITEMS) {
        float v = rowp[idx];
        if (better(v, idx, thrV, thrI)) {
          int slot = atomicAdd(&cnt, 1);
          qv[slot] = v; qi[slot] = idx;
        }
      }
    }
    __syncthreads();
    int n = cnt;
    for (int s = 0; s < n; ++s) {
      float cv = qv[s];
      int   ci = qi[s];
      bool enter = better(cv, ci, topv[TOPK - 1], topi[TOPK - 1]); // uniform
      float myv = 0.f; int myi = 0;
      bool myB = true, prevB = true;
      if (enter && tid < TOPK) {
        myv = topv[tid]; myi = topi[tid];
        myB = better(myv, myi, cv, ci);
        prevB = (tid == 0) ? true : better(topv[tid - 1], topi[tid - 1], cv, ci);
      }
      __syncthreads();
      if (enter && tid < TOPK && !myB) {
        if (tid + 1 < TOPK) { topv[tid + 1] = myv; topi[tid + 1] = myi; }
        if (prevB)          { topv[tid] = cv; topi[tid] = ci; }
      }
      __syncthreads();
    }
    __syncthreads();
    if (tid == 0) cnt = 0;
    __syncthreads();
  }

  // ---- write outputs: logits then ids (ids < 1e7 are exact in f32)
  if (tid < TOPK) {
    out[q * TOPK + tid] = topv[tid];
    out[NQ * TOPK + q * TOPK + tid] = (float)ids[topi[tid]];
  }
}

// ---------------------------------------------------------------------------
extern "C" void kernel_launch(void* const* d_in, const int* in_sizes, int n_in,
                              void* d_out, int out_size, void* d_ws, size_t ws_size,
                              hipStream_t stream) {
  const float* Q   = (const float*)d_in[0];   // (128,128)
  const float* Tt  = (const float*)d_in[1];   // (128,1e6)
  const int*   ids = (const int*)d_in[2];     // (1e6,)
  float* out = (float*)d_out;                 // 128*100 logits, then 128*100 ids
  float* S   = (float*)d_ws;                  // 128*1e6 fp32 scores (512 MB)

  size_t shmem = (size_t)SBF_BYTES + (size_t)2 * NQ * AROW * sizeof(unsigned short);
  mips_gemm_scores_kernel<<<NITEMS / 64, 128, shmem, stream>>>(Q, Tt, S);
  mips_topk_kernel<<<NQ, BT2, 0, stream>>>(S, ids, out);
}
